// MultiheadAttention_34591666602455
// MI455X (gfx1250) — matrix operand order repaired and checked by reading, not compiled, not run on hardware
//
#include <hip/hip_runtime.h>
#include <stdint.h>

typedef __attribute__((ext_vector_type(16))) _Float16 v16h;
typedef __attribute__((ext_vector_type(8)))  float    v8f;
typedef __attribute__((ext_vector_type(4)))  unsigned int u32x4;

#define B_  4
#define L_  2048
#define D_  1024
#define H_  16
#define HD_ 64
#define BL_ (B_*L_)
#define NEGM (-1e9f)

union AFrag { v16h v; u32x4 q[2]; };
union BFrag { v16h v; u32x4 q[2]; };

__device__ inline v8f wmma16(v16h a, v16h b, v8f c) {
  // D = A(16x32 f16) * B(32x16 f16) + C(16x16 f32)
  return __builtin_amdgcn_wmma_f32_16x16x32_f16(false, a, false, b, (short)0, c, false, false);
}

// ---------------------------------------------------------------------------
// 1) pre-LayerNorm: h = LN(x)*g_pre + b_pre, emitted as f16 (GEMM A operand)
// ---------------------------------------------------------------------------
__global__ __launch_bounds__(256) void ln_pre_kernel(
    const float* __restrict__ x, const float* __restrict__ g,
    const float* __restrict__ b, _Float16* __restrict__ h16)
{
  __shared__ float sa[256], sb[256];
  int row = blockIdx.x, t = threadIdx.x;
  const float* xr = x + (size_t)row * D_;
  float v[4], s = 0.f, ss = 0.f;
#pragma unroll
  for (int j = 0; j < 4; ++j) { v[j] = xr[t + j*256]; s += v[j]; ss += v[j]*v[j]; }
  sa[t] = s; sb[t] = ss; __syncthreads();
  for (int k = 128; k > 0; k >>= 1) {
    if (t < k) { sa[t] += sa[t+k]; sb[t] += sb[t+k]; }
    __syncthreads();
  }
  float mu  = sa[0] * (1.f/D_);
  float var = sb[0] * (1.f/D_) - mu*mu;
  float rs  = rsqrtf(var + 1e-5f);
  _Float16* hr = h16 + (size_t)row * D_;
#pragma unroll
  for (int j = 0; j < 4; ++j) {
    int i = t + j*256;
    hr[i] = (_Float16)((v[j]-mu)*rs*g[i] + b[i]);
  }
}

// ---------------------------------------------------------------------------
// 2) f32 -> f16 weight conversion
// ---------------------------------------------------------------------------
__global__ __launch_bounds__(256) void cvt_kernel(
    const float* __restrict__ s, _Float16* __restrict__ d, int n)
{
  for (int i = blockIdx.x*256 + threadIdx.x; i < n; i += gridDim.x*256)
    d[i] = (_Float16)s[i];
}

// ---------------------------------------------------------------------------
// 3) WMMA GEMM: C[M=8192,N=1024] = A(f16,row-major) @ W^T (W is [N,K] f16) + bias
//    Block tile 128x64, 8 waves of 32x32 (2x2 WMMA tiles), K-step 32, LDS staged.
//    mode 0/1/2: scatter f16 into Q/K (B,H,L,HD) or V^T (B,H,HD,L)
//    mode 3:     fp32 row-major into Of32
// ---------------------------------------------------------------------------
__device__ inline void store_tile(v8f c, int m0, int n0, int g, int ln,
                                  const float* __restrict__ bias, int mode,
                                  _Float16* __restrict__ Of16, float* __restrict__ Of32)
{
  int n = n0 + ln;
  float bv = bias[n];
#pragma unroll
  for (int r = 0; r < 8; ++r) {
    int m = m0 + r + 8*g;                 // C layout: lane=n, VGPR r -> M = r + 8*(lane>=16)
    float val = c[r] + bv;
    if (mode == 3) {
      Of32[(size_t)m * D_ + n] = val;
    } else {
      int bb = m >> 11, l = m & (L_-1);
      int hh = n >> 6,  hd = n & 63;
      if (mode == 2)   // V transposed: (B,H,HD,L)
        Of16[(((size_t)bb*H_ + hh)*HD_ + hd)*L_ + l] = (_Float16)val;
      else             // Q / K: (B,H,L,HD)
        Of16[(((size_t)bb*H_ + hh)*L_ + l)*HD_ + hd] = (_Float16)val;
    }
  }
}

__global__ __launch_bounds__(256) void gemm_kernel(
    const _Float16* __restrict__ A,
    const _Float16* __restrict__ W0, const _Float16* __restrict__ W1,
    const _Float16* __restrict__ W2,
    const float* __restrict__ b0, const float* __restrict__ b1,
    const float* __restrict__ b2,
    _Float16* __restrict__ O0, _Float16* __restrict__ O1, _Float16* __restrict__ O2,
    float* __restrict__ Of32, int mode_base)
{
  __shared__ __align__(16) _Float16 At[128*40];   // pitch 40 halves keeps 16B align
  __shared__ __align__(16) _Float16 Bt[64*40];
  int t = threadIdx.x, lane = t & 31, w = t >> 5;
  int wm = w >> 1, wn = w & 1;
  int g = lane >> 4, ln = lane & 15;
  int bm0 = blockIdx.x * 128, bn0 = blockIdx.y * 64;
  int mode = mode_base + blockIdx.z;
  const _Float16* W    = (mode==1) ? W1 : (mode==2) ? W2 : W0;
  const float*    bias = (mode==1) ? b1 : (mode==2) ? b2 : b0;
  _Float16*       Of16 = (mode==1) ? O1 : (mode==2) ? O2 : O0;

  int ar = t >> 1, ac = (t & 1) * 16;   // A tile: 128 rows x 32 halves
  int br = t >> 2, bc = (t & 3) * 8;    // W tile:  64 rows x 32 halves

  v8f c00 = {}, c01 = {}, c10 = {}, c11 = {};

  for (int k0 = 0; k0 < D_; k0 += 32) {
    const _Float16* asrc = A + (size_t)(bm0 + ar)*D_ + k0 + ac;
    *(u32x4*)(At + ar*40 + ac)     = *(const u32x4*)(asrc);
    *(u32x4*)(At + ar*40 + ac + 8) = *(const u32x4*)(asrc + 8);
    const _Float16* bsrc = W + (size_t)(bn0 + br)*D_ + k0 + bc;
    *(u32x4*)(Bt + br*40 + bc) = *(const u32x4*)(bsrc);
    __syncthreads();

    AFrag a0, a1; BFrag f0, f1;
    const _Float16* ap0 = At + (wm*32 + ln)*40;
    a0.q[0] = *(const u32x4*)(ap0 + 8*g);
    a0.q[1] = *(const u32x4*)(ap0 + 16 + 8*g);
    const _Float16* ap1 = At + (wm*32 + 16 + ln)*40;
    a1.q[0] = *(const u32x4*)(ap1 + 8*g);
    a1.q[1] = *(const u32x4*)(ap1 + 16 + 8*g);
    const _Float16* bp0 = Bt + (wn*32 + ln)*40 + 8*g;
    f0.q[0] = *(const u32x4*)(bp0);
    f0.q[1] = *(const u32x4*)(bp0 + 16);
    const _Float16* bp1 = Bt + (wn*32 + 16 + ln)*40 + 8*g;
    f1.q[0] = *(const u32x4*)(bp1);
    f1.q[1] = *(const u32x4*)(bp1 + 16);

    c00 = wmma16(a0.v, f0.v, c00);
    c01 = wmma16(a0.v, f1.v, c01);
    c10 = wmma16(a1.v, f0.v, c10);
    c11 = wmma16(a1.v, f1.v, c11);
    __syncthreads();
  }

  store_tile(c00, bm0 + wm*32,      bn0 + wn*32,      g, ln, bias, mode, Of16, Of32);
  store_tile(c01, bm0 + wm*32,      bn0 + wn*32 + 16, g, ln, bias, mode, Of16, Of32);
  store_tile(c10, bm0 + wm*32 + 16, bn0 + wn*32,      g, ln, bias, mode, Of16, Of32);
  store_tile(c11, bm0 + wm*32 + 16, bn0 + wn*32 + 16, g, ln, bias, mode, Of16, Of32);
}

// ---------------------------------------------------------------------------
// 4) Flash attention. Each wave owns 16 query rows; keys in chunks of 32.
//    scores = Q K^T / 8 (+ causal/pad mask), online softmax, O += P V.
// ---------------------------------------------------------------------------
__global__ __launch_bounds__(256) void attn_kernel(
    const _Float16* __restrict__ Qf, const _Float16* __restrict__ Kf,
    const _Float16* __restrict__ Vt, const unsigned char* __restrict__ pad,
    _Float16* __restrict__ attnO)
{
  __shared__ __align__(16) _Float16 Pbuf[8][16*40];
  int t = threadIdx.x, lane = t & 31, w = t >> 5;
  int g = lane >> 4, ln = lane & 15;
  int bh = blockIdx.y, b = bh >> 4, hh = bh & 15;
  int l0 = blockIdx.x * 128 + w * 16;
  const _Float16* Qb = Qf + (size_t)bh * L_ * HD_;
  const _Float16* Kb = Kf + (size_t)bh * L_ * HD_;
  const _Float16* Vb = Vt + (size_t)bh * HD_ * L_;
  const unsigned char* padb = pad + (size_t)b * L_;

  AFrag aq0, aq1;                                // Q rows l0..l0+15, K-dim 64
  const _Float16* qrow = Qb + (size_t)(l0 + ln) * HD_;
  aq0.q[0] = *(const u32x4*)(qrow + 8*g);
  aq0.q[1] = *(const u32x4*)(qrow + 16 + 8*g);
  aq1.q[0] = *(const u32x4*)(qrow + 32 + 8*g);
  aq1.q[1] = *(const u32x4*)(qrow + 48 + 8*g);

  v8f o0 = {}, o1 = {}, o2 = {}, o3 = {};
  float mprev[8], lsum[8];
#pragma unroll
  for (int r = 0; r < 8; ++r) { mprev[r] = -3.0e38f; lsum[r] = 0.f; }

  _Float16* P = Pbuf[w];

  for (int kb = 0; kb < l0 + 16; kb += 32) {
    // K B-frags: 2 key tiles x 2 contraction chunks (HD=64)
    BFrag k00, k01, k10, k11;
    const _Float16* kr0 = Kb + (size_t)(kb + ln) * HD_;
    k00.q[0] = *(const u32x4*)(kr0 + 8*g);
    k00.q[1] = *(const u32x4*)(kr0 + 8*g + 16);
    k01.q[0] = *(const u32x4*)(kr0 + 32 + 8*g);
    k01.q[1] = *(const u32x4*)(kr0 + 32 + 8*g + 16);
    const _Float16* kr1 = Kb + (size_t)(kb + 16 + ln) * HD_;
    k10.q[0] = *(const u32x4*)(kr1 + 8*g);
    k10.q[1] = *(const u32x4*)(kr1 + 8*g + 16);
    k11.q[0] = *(const u32x4*)(kr1 + 32 + 8*g);
    k11.q[1] = *(const u32x4*)(kr1 + 32 + 8*g + 16);

    v8f s0 = {}, s1 = {};
    s0 = wmma16(aq0.v, k00.v, s0);
    s0 = wmma16(aq1.v, k01.v, s0);
    s1 = wmma16(aq0.v, k10.v, s1);
    s1 = wmma16(aq1.v, k11.v, s1);

    int kidx0 = kb + ln, kidx1 = kb + 16 + ln;
    bool p0m = (padb[kidx0] != 0);
    bool p1m = (padb[kidx1] != 0);

#pragma unroll
    for (int r = 0; r < 8; ++r) {
      int l = l0 + r + 8*g;
      float v0 = (p0m || kidx0 > l) ? NEGM : s0[r] * 0.125f;
      float v1 = (p1m || kidx1 > l) ? NEGM : s1[r] * 0.125f;
      // row max across the 16 lanes of this half (C layout: lane = key col)
      float mx = fmaxf(v0, v1);
      mx = fmaxf(mx, __shfl_xor(mx, 1));
      mx = fmaxf(mx, __shfl_xor(mx, 2));
      mx = fmaxf(mx, __shfl_xor(mx, 4));
      mx = fmaxf(mx, __shfl_xor(mx, 8));
      float mi = fmaxf(mprev[r], mx);
      float alpha = __expf(mprev[r] - mi);
      float e0 = __expf(v0 - mi);
      float e1 = __expf(v1 - mi);
      float rsum = e0 + e1;
      rsum += __shfl_xor(rsum, 1);
      rsum += __shfl_xor(rsum, 2);
      rsum += __shfl_xor(rsum, 4);
      rsum += __shfl_xor(rsum, 8);
      lsum[r]  = lsum[r] * alpha + rsum;
      mprev[r] = mi;
      o0[r] *= alpha; o1[r] *= alpha; o2[r] *= alpha; o3[r] *= alpha;
      P[(r + 8*g)*40 + ln]      = (_Float16)e0;   // C-layout -> LDS (row-major 16x32)
      P[(r + 8*g)*40 + 16 + ln] = (_Float16)e1;
    }

    // P back as A-frag (LDS ds ops are in-order within a wave)
    AFrag ap;
    const _Float16* prow = P + ln*40;
    ap.q[0] = *(const u32x4*)(prow + 8*g);
    ap.q[1] = *(const u32x4*)(prow + 16 + 8*g);

    // V^T B-frags: 4 hd tiles, contraction over 32 keys (contiguous in Vt rows)
    BFrag bv0, bv1, bv2, bv3;
    const _Float16* vr0 = Vb + (size_t)(ln)      * L_ + kb;
    bv0.q[0] = *(const u32x4*)(vr0 + 8*g);  bv0.q[1] = *(const u32x4*)(vr0 + 8*g + 16);
    const _Float16* vr1 = Vb + (size_t)(16 + ln) * L_ + kb;
    bv1.q[0] = *(const u32x4*)(vr1 + 8*g);  bv1.q[1] = *(const u32x4*)(vr1 + 8*g + 16);
    const _Float16* vr2 = Vb + (size_t)(32 + ln) * L_ + kb;
    bv2.q[0] = *(const u32x4*)(vr2 + 8*g);  bv2.q[1] = *(const u32x4*)(vr2 + 8*g + 16);
    const _Float16* vr3 = Vb + (size_t)(48 + ln) * L_ + kb;
    bv3.q[0] = *(const u32x4*)(vr3 + 8*g);  bv3.q[1] = *(const u32x4*)(vr3 + 8*g + 16);

    o0 = wmma16(ap.v, bv0.v, o0);
    o1 = wmma16(ap.v, bv1.v, o1);
    o2 = wmma16(ap.v, bv2.v, o2);
    o3 = wmma16(ap.v, bv3.v, o3);
  }

#pragma unroll
  for (int r = 0; r < 8; ++r) {
    float inv = 1.f / lsum[r];
    int l = l0 + r + 8*g;
    size_t base = ((size_t)(b * L_ + l)) * D_ + (size_t)hh * HD_ + ln;
    attnO[base +  0] = (_Float16)(o0[r] * inv);
    attnO[base + 16] = (_Float16)(o1[r] * inv);
    attnO[base + 32] = (_Float16)(o2[r] * inv);
    attnO[base + 48] = (_Float16)(o3[r] * inv);
  }
}

// ---------------------------------------------------------------------------
// 5) final: out = LN( y + LN(x)*g_pre + b_pre ) * g_ln + b_ln  (in-place on y)
// ---------------------------------------------------------------------------
__global__ __launch_bounds__(256) void ln_post_kernel(
    const float* __restrict__ x,
    const float* __restrict__ gpre, const float* __restrict__ bpre,
    const float* __restrict__ gln,  const float* __restrict__ bln,
    float* __restrict__ y)
{
  __shared__ float sa[256], sb[256];
  int row = blockIdx.x, t = threadIdx.x;
  const float* xr = x + (size_t)row * D_;
  float* yr = y + (size_t)row * D_;
  float xv[4], s = 0.f, ss = 0.f;
#pragma unroll
  for (int j = 0; j < 4; ++j) { xv[j] = xr[t + j*256]; s += xv[j]; ss += xv[j]*xv[j]; }
  sa[t] = s; sb[t] = ss; __syncthreads();
  for (int k = 128; k > 0; k >>= 1) {
    if (t < k) { sa[t] += sa[t+k]; sb[t] += sb[t+k]; }
    __syncthreads();
  }
  float mu1 = sa[0] * (1.f/D_);
  float rs1 = rsqrtf(sb[0] * (1.f/D_) - mu1*mu1 + 1e-5f);
  __syncthreads();

  float rv[4]; s = 0.f; ss = 0.f;
#pragma unroll
  for (int j = 0; j < 4; ++j) {
    int i = t + j*256;
    float h = (xv[j]-mu1)*rs1*gpre[i] + bpre[i];
    rv[j] = yr[i] + h;
    s += rv[j]; ss += rv[j]*rv[j];
  }
  sa[t] = s; sb[t] = ss; __syncthreads();
  for (int k = 128; k > 0; k >>= 1) {
    if (t < k) { sa[t] += sa[t+k]; sb[t] += sb[t+k]; }
    __syncthreads();
  }
  float mu2 = sa[0] * (1.f/D_);
  float rs2 = rsqrtf(sb[0] * (1.f/D_) - mu2*mu2 + 1e-5f);
#pragma unroll
  for (int j = 0; j < 4; ++j) {
    int i = t + j*256;
    yr[i] = (rv[j]-mu2)*rs2*gln[i] + bln[i];
  }
}

// ---------------------------------------------------------------------------
extern "C" void kernel_launch(void* const* d_in, const int* in_sizes, int n_in,
                              void* d_out, int out_size, void* d_ws, size_t ws_size,
                              hipStream_t stream)
{
  const float* x            = (const float*)d_in[0];
  const unsigned char* pad  = (const unsigned char*)d_in[1];   // (B,L) bool
  // d_in[2] = causal_mask: semantics reproduced analytically in-kernel
  const float* Wq = (const float*)d_in[3];  const float* bq = (const float*)d_in[4];
  const float* Wk = (const float*)d_in[5];  const float* bk = (const float*)d_in[6];
  const float* Wv = (const float*)d_in[7];  const float* bv = (const float*)d_in[8];
  const float* Wo = (const float*)d_in[9];  const float* bo = (const float*)d_in[10];
  const float* gpre = (const float*)d_in[11]; const float* bpre = (const float*)d_in[12];
  const float* gln  = (const float*)d_in[13]; const float* bln  = (const float*)d_in[14];
  float* out = (float*)d_out;

  char* ws = (char*)d_ws;
  size_t off = 0;
  auto alloc = [&](size_t bytes) -> void* {
    void* p = ws + off; off += (bytes + 255) & ~(size_t)255; return p;
  };
  _Float16* h16  = (_Float16*)alloc((size_t)BL_ * D_ * 2);
  _Float16* Wq16 = (_Float16*)alloc((size_t)D_ * D_ * 2);
  _Float16* Wk16 = (_Float16*)alloc((size_t)D_ * D_ * 2);
  _Float16* Wv16 = (_Float16*)alloc((size_t)D_ * D_ * 2);
  _Float16* Wo16 = (_Float16*)alloc((size_t)D_ * D_ * 2);
  _Float16* Qf   = (_Float16*)alloc((size_t)BL_ * D_ * 2);
  _Float16* Kf   = (_Float16*)alloc((size_t)BL_ * D_ * 2);
  _Float16* Vt   = (_Float16*)alloc((size_t)BL_ * D_ * 2);
  _Float16* At16 = (_Float16*)alloc((size_t)BL_ * D_ * 2);

  ln_pre_kernel<<<BL_, 256, 0, stream>>>(x, gpre, bpre, h16);
  cvt_kernel<<<1024, 256, 0, stream>>>(Wq, Wq16, D_*D_);
  cvt_kernel<<<1024, 256, 0, stream>>>(Wk, Wk16, D_*D_);
  cvt_kernel<<<1024, 256, 0, stream>>>(Wv, Wv16, D_*D_);
  cvt_kernel<<<1024, 256, 0, stream>>>(Wo, Wo16, D_*D_);
  // fused Q/K/V projection: grid.z picks weight/bias/output + layout
  gemm_kernel<<<dim3(BL_/128, D_/64, 3), 256, 0, stream>>>(
      h16, Wq16, Wk16, Wv16, bq, bk, bv, Qf, Kf, Vt, nullptr, 0);
  attn_kernel<<<dim3(L_/128, B_*H_), 256, 0, stream>>>(Qf, Kf, Vt, pad, At16);
  // output projection -> fp32 into d_out (mode 3)
  gemm_kernel<<<dim3(BL_/128, D_/64, 1), 256, 0, stream>>>(
      At16, Wo16, nullptr, nullptr, bo, nullptr, nullptr,
      nullptr, nullptr, nullptr, out, 3);
  ln_post_kernel<<<BL_, 256, 0, stream>>>(x, gpre, bpre, gln, bln, out);
}
